// UnderlyingTwoDimensionalGRU_13159779795692
// MI455X (gfx1250) — compile-verified, matrix operand order checked by reading, not run on hardware
//
#include <hip/hip_runtime.h>
#include <hip/hip_bf16.h>

typedef __attribute__((ext_vector_type(16))) __bf16 v16bf;
typedef __attribute__((ext_vector_type(8)))  float  v8f;

#define B_   32
#define PR_  48
#define PC_  48
#define E_   256
#define H_   256
#define GH_  512
#define NG_  1536                    // 3*GH
#define Q_   (B_ * PR_ * PC_)       // 73728 positions, cell-major: q = (r*PC+c)*B + b

union Frag { unsigned short u[16]; v16bf v; };

__device__ __forceinline__ unsigned short f2bf(float f) {
  unsigned u = __float_as_uint(f);
  unsigned r = 0x7FFFu + ((u >> 16) & 1u);
  return (unsigned short)((u + r) >> 16);
}
__device__ __forceinline__ float bf2f(unsigned short h) {
  return __uint_as_float(((unsigned)h) << 16);
}
__device__ __forceinline__ float sigmoidf_(float x) { return 1.f / (1.f + __expf(-x)); }

__device__ __forceinline__ v8f wmma_bf16(v16bf a, v16bf b, v8f c) {
  return __builtin_amdgcn_wmma_f32_16x16x32_bf16(false, a, false, b, (short)0, c, false, false);
}

// 16 bf16 elems of a fragment: 8 contiguous at p[kb..kb+7], 8 at p[16+kb..16+kb+7]
__device__ __forceinline__ v16bf load_frag16(const unsigned short* p, int kb) {
  Frag t;
  #pragma unroll
  for (int j = 0; j < 8; ++j) { t.u[j] = p[kb + j]; t.u[8 + j] = p[16 + kb + j]; }
  return t.v;
}
__device__ __forceinline__ v16bf zero_frag() {
  Frag t;
  #pragma unroll
  for (int j = 0; j < 16; ++j) t.u[j] = 0;
  return t.v;
}
// B-fragment of g = h @ W^T at tile (n0,k0): B[k][n] = W[n0+n][k0+k], W row-major bf16 [N][ldk]
__device__ __forceinline__ v16bf load_b(const unsigned short* __restrict__ W, int ldk,
                                        int n0, int k0, int nloc, int kb) {
  return load_frag16(W + (size_t)(n0 + nloc) * ldk + k0, kb);
}

// ---- prep: weights fp32->bf16, x -> bf16 cell-major copy, progress-flag init ----
__global__ __launch_bounds__(256) void prep_convert(
    const float* __restrict__ x,
    const float* __restrict__ Wih, const float* __restrict__ Whh,
    const float* __restrict__ Wcomp, const float* __restrict__ Wres,
    unsigned short* __restrict__ wih, unsigned short* __restrict__ whh,
    unsigned short* __restrict__ wcomp, unsigned short* __restrict__ wres,
    unsigned short* __restrict__ xbf,     // [Q][E] bf16, q = (r*PC+c)*B + b
    int* __restrict__ progress) {
  const int i = blockIdx.x * blockDim.x + threadIdx.x;
  if (i < 64) progress[i] = 0;
  const int n1 = NG_ * E_;
  const int n2 = n1 + NG_ * GH_;
  const int n3 = n2 + H_ * GH_;
  const int n4 = n3 + H_ * E_;
  if (i < n1)      wih[i]        = f2bf(Wih[i]);
  else if (i < n2) whh[i - n1]   = f2bf(Whh[i - n1]);
  else if (i < n3) wcomp[i - n2] = f2bf(Wcomp[i - n2]);
  else if (i < n4) wres[i - n3]  = f2bf(Wres[i - n3]);
  else if (i < n4 + Q_ * E_) {
    const int idx = i - n4;
    const int q  = idx >> 8;          // E_ = 256
    const int e  = idx & (E_ - 1);
    const int bb = q & (B_ - 1);      // B_ = 32
    const int ci = q >> 5;
    const int r  = ci / PC_;
    const int c  = ci % PC_;
    xbf[idx] = f2bf(x[(((size_t)bb * PR_ + r) * PC_ + c) * E_ + e]);
  }
}

// ---- fully parallel precompute into f32 TILE-MAJOR layout ----
// xg_t[(mt*NG + col)*16 + m], res_t[(mt*H + col)*16 + m];  mt over cell-major q-space.
// grid: 4608 m-tiles x 14 segments; segs 0..11 -> xg (96 n-tiles), 12..13 -> res (16 n-tiles)
__global__ __launch_bounds__(256) void xg_res_gemm(
    const unsigned short* __restrict__ xbf,   // [Q][256] bf16
    const float* __restrict__ b_ih, const float* __restrict__ b_hh,
    const float* __restrict__ b_comp, const float* __restrict__ b_res,
    const unsigned short* __restrict__ Wih,   // [1536][256] bf16
    const unsigned short* __restrict__ Wres,  // [256][256]  bf16
    float* __restrict__ xg,                   // tile-major f32
    float* __restrict__ resb) {               // tile-major f32
  const int bid  = blockIdx.x;
  const int mt   = bid / 14;
  const int seg  = bid % 14;
  const int tid  = threadIdx.x;
  const int wave = tid >> 5;
  const int lane = tid & 31;
  const int nloc = lane & 15;
  const int kb   = (lane >> 4) * 8;
  const int mrb  = (lane >> 4) * 8;

  const bool isxg = (seg < 12);
  const int n0   = isxg ? (seg * 8 + wave) * 16 : ((seg - 12) * 8 + wave) * 16;
  const int ncol = n0 + nloc;
  const unsigned short* W = isxg ? Wih : Wres;   // both ldk = E_

  float bias;
  if (isxg) bias = b_ih[ncol] + (ncol < 2 * GH_ ? b_hh[ncol] : 0.f);  // r,z: both; n: b_ih only
  else      bias = b_res[ncol] + b_comp[ncol];
  v8f acc;
  #pragma unroll
  for (int i = 0; i < 8; ++i) acc[i] = bias;

  const unsigned short* xp = xbf + (size_t)(mt * 16 + nloc) * E_;   // A-fragment row (bf16)
  #pragma unroll 1
  for (int k0 = 0; k0 < E_; k0 += 32) {
    acc = wmma_bf16(load_frag16(xp + k0, kb), load_b(W, E_, n0, k0, nloc, kb), acc);
  }
  // D-layout-native store: lane's 8 values are contiguous in tile-major memory
  if (isxg) {
    float* d = xg + ((size_t)mt * NG_ + ncol) * 16 + mrb;
    #pragma unroll
    for (int i = 0; i < 8; ++i) d[i] = acc[i];
  } else {
    float* d = resb + ((size_t)mt * H_ + ncol) * 16 + mrb;
    #pragma unroll
    for (int i = 0; i < 8; ++i) d[i] = acc[i];
  }
}

// ---------------- persistent wavefront 2D-GRU kernel (recurrent path only) ----------------
__global__ __launch_bounds__(256) void gru2d_wavefront(
    const float* __restrict__ b_hh,
    const float* __restrict__ gamma, const float* __restrict__ beta,
    const unsigned short* __restrict__ Whh,    // [1536][512] bf16
    const unsigned short* __restrict__ Wcomp,  // [256][512]  bf16
    const float* __restrict__ xg,              // tile-major f32 (biases folded)
    const float* __restrict__ resb,            // tile-major f32 (biases folded)
    unsigned short* __restrict__ hbuf,         // [Q][256] bf16 hidden-state mirror (q-indexed)
    float* __restrict__ out, int* __restrict__ progress) {
  __shared__ unsigned short sHnew[B_][GH_ + 2];
  __shared__ unsigned short sHc[B_][H_ + 2];
  __shared__ float sMu[B_], sRs[B_];

  const int r    = blockIdx.x;       // one WG per grid row, 48 pipelined WGs
  const int tid  = threadIdx.x;
  const int wave = tid >> 5;         // wave32
  const int lane = tid & 31;
  const int nloc = lane & 15;
  const int kb   = (lane >> 4) * 8;
  const int mrb  = (lane >> 4) * 8;  // D-tile: M = i + 8*(lane>>4)

  #pragma unroll 1
  for (int c = 0; c < PC_; ++c) {
    if (r > 0) {
      if (tid == 0) {
        while (__atomic_load_n(progress + (r - 1), __ATOMIC_ACQUIRE) <= c)
          __builtin_amdgcn_s_sleep(1);
      }
      __syncthreads();
    }
    const int ci  = r * PC_ + c;             // this cell
    const int ciu = (r - 1) * PC_ + c;       // cell above
    const int cil = ci - 1;                  // cell to the left

    // ---- GEMM1: gates = xg (f32 tile-major) + h_prev @ Whh^T ; h_new -> LDS ----
    #pragma unroll 1
    for (int cc = 0; cc < 4; ++cc) {
      const int u0   = (wave + 8 * cc) * 16;
      const int ncol = u0 + nloc;
      v8f racc[2], zacc[2], xnacc[2], hnacc[2];
      const float hnb = b_hh[2 * GH_ + ncol];
      #pragma unroll
      for (int rt = 0; rt < 2; ++rt) {
        const size_t mt = (size_t)(ci * 2 + rt);
        const float* gr = xg + (mt * NG_ + ncol) * 16 + mrb;            // 8 contiguous f32
        const float* gz = xg + (mt * NG_ + GH_ + ncol) * 16 + mrb;
        const float* gn = xg + (mt * NG_ + 2 * GH_ + ncol) * 16 + mrb;
        #pragma unroll
        for (int i = 0; i < 8; ++i) {
          racc[rt][i]  = gr[i];
          zacc[rt][i]  = gz[i];
          xnacc[rt][i] = gn[i];
          hnacc[rt][i] = hnb;
        }
      }
      #pragma unroll 1
      for (int k0 = 0; k0 < GH_; k0 += 32) {
        // B-fragments shared across both batch row-tiles
        const v16bf br = load_b(Whh, GH_, u0,           k0, nloc, kb);
        const v16bf bz = load_b(Whh, GH_, GH_ + u0,     k0, nloc, kb);
        const v16bf bn = load_b(Whh, GH_, 2 * GH_ + u0, k0, nloc, kb);
        #pragma unroll
        for (int rt = 0; rt < 2; ++rt) {
          const int arow = rt * 16 + nloc;
          v16bf a;
          if (k0 < H_) {   // K 0..255: hidden state of cell above
            a = (r > 0) ? load_frag16(hbuf + ((size_t)ciu * B_ + arow) * H_ + k0, kb)
                        : zero_frag();
          } else {         // K 256..511: hidden state of cell to the left
            a = (c > 0) ? load_frag16(hbuf + ((size_t)cil * B_ + arow) * H_ + (k0 - H_), kb)
                        : zero_frag();
          }
          racc[rt]  = wmma_bf16(a, br, racc[rt]);
          zacc[rt]  = wmma_bf16(a, bz, zacc[rt]);
          hnacc[rt] = wmma_bf16(a, bn, hnacc[rt]);
        }
      }
      #pragma unroll
      for (int rt = 0; rt < 2; ++rt) {
        #pragma unroll
        for (int i = 0; i < 8; ++i) {
          const int bb = rt * 16 + mrb + i;
          float hp = 0.f;
          if (ncol < H_) {
            if (r > 0) hp = bf2f(hbuf[((size_t)ciu * B_ + bb) * H_ + ncol]);
          } else {
            if (c > 0) hp = bf2f(hbuf[((size_t)cil * B_ + bb) * H_ + (ncol - H_)]);
          }
          const float rv = sigmoidf_(racc[rt][i]);
          const float zv = sigmoidf_(zacc[rt][i]);
          const float nv = tanhf(xnacc[rt][i] + rv * hnacc[rt][i]);  // r gates only h-path
          sHnew[bb][ncol] = f2bf((1.f - zv) * nv + zv * hp);
        }
      }
    }
    __syncthreads();

    // ---- GEMM2: h_c = h_new @ Wcomp^T + res (f32 tile-major, biases folded) ----
    #pragma unroll 1
    for (int half = 0; half < 2; ++half) {
      const int n0   = (wave + 8 * half) * 16;
      const int ncol = n0 + nloc;
      v8f acc[2];
      #pragma unroll
      for (int rt = 0; rt < 2; ++rt) {
        const float* rs = resb + ((size_t)(ci * 2 + rt) * H_ + ncol) * 16 + mrb;
        #pragma unroll
        for (int i = 0; i < 8; ++i) acc[rt][i] = rs[i];
      }
      #pragma unroll 1
      for (int k0 = 0; k0 < GH_; k0 += 32) {
        const v16bf bw = load_b(Wcomp, GH_, n0, k0, nloc, kb);
        #pragma unroll
        for (int rt = 0; rt < 2; ++rt) {
          const v16bf a = load_frag16(&sHnew[rt * 16 + nloc][0] + k0, kb);
          acc[rt] = wmma_bf16(a, bw, acc[rt]);
        }
      }
      #pragma unroll
      for (int rt = 0; rt < 2; ++rt) {
        #pragma unroll
        for (int i = 0; i < 8; ++i) sHc[rt * 16 + mrb + i][ncol] = f2bf(acc[rt][i]);
      }
    }
    __syncthreads();

    // ---- LayerNorm over H per batch row ----
    if (tid < B_) {
      float s = 0.f, s2 = 0.f;
      for (int j = 0; j < H_; ++j) {
        const float v = bf2f(sHc[tid][j]);
        s += v; s2 += v * v;
      }
      const float mu  = s * (1.f / H_);
      const float var = s2 * (1.f / H_) - mu * mu;
      sMu[tid] = mu;
      sRs[tid] = __frsqrt_rn(var + 1e-5f);
    }
    __syncthreads();
    {
      const int bb = tid >> 3;
      const int j0 = (tid & 7) * 32;
      const float mu = sMu[bb], rs = sRs[bb];
      float* op = out + (((size_t)bb * PR_ + r) * PC_ + c) * H_;   // reference layout
      unsigned short* hb = hbuf + ((size_t)ci * B_ + bb) * H_;     // cell-major mirror
      #pragma unroll 1
      for (int e = 0; e < 32; ++e) {
        const int j = j0 + e;
        const float v = (bf2f(sHc[bb][j]) - mu) * rs * gamma[j] + beta[j];
        op[j] = v;
        hb[j] = f2bf(v);
      }
    }

    __threadfence();
    __syncthreads();
    if (tid == 0)
      __atomic_store_n(progress + r, c + 1, __ATOMIC_RELEASE);
  }
}

// ---------------- h_final extraction ----------------
__global__ __launch_bounds__(256) void finalize_hfinal(const float* __restrict__ out,
                                                       float* __restrict__ hfin) {
  const int i = blockIdx.x * blockDim.x + threadIdx.x;
  if (i < B_ * H_) {
    const int b = i >> 8, h = i & (H_ - 1);
    hfin[i] = out[(((size_t)b * PR_ + (PR_ - 1)) * PC_ + (PC_ - 1)) * H_ + h];
  }
}

extern "C" void kernel_launch(void* const* d_in, const int* in_sizes, int n_in,
                              void* d_out, int out_size, void* d_ws, size_t ws_size,
                              hipStream_t stream) {
  const float* x      = (const float*)d_in[0];
  const float* W_ih   = (const float*)d_in[1];
  const float* W_hh   = (const float*)d_in[2];
  const float* b_ih   = (const float*)d_in[3];
  const float* b_hh   = (const float*)d_in[4];
  const float* W_comp = (const float*)d_in[5];
  const float* b_comp = (const float*)d_in[6];
  const float* W_res  = (const float*)d_in[7];
  const float* b_res  = (const float*)d_in[8];
  const float* gamma  = (const float*)d_in[9];
  const float* beta   = (const float*)d_in[10];

  char* ws = (char*)d_ws;
  size_t off = 0;
  int* progress = (int*)ws;                              off += 256;
  unsigned short* wih   = (unsigned short*)(ws + off);   off += (size_t)NG_ * E_  * 2;
  unsigned short* whh   = (unsigned short*)(ws + off);   off += (size_t)NG_ * GH_ * 2;
  unsigned short* wcomp = (unsigned short*)(ws + off);   off += (size_t)H_  * GH_ * 2;
  unsigned short* wres  = (unsigned short*)(ws + off);   off += (size_t)H_  * E_  * 2;
  unsigned short* xbf   = (unsigned short*)(ws + off);   off += (size_t)Q_  * E_  * 2;  //  37.7 MB
  float*          xg    = (float*)(ws + off);            off += (size_t)Q_  * NG_ * 4;  // 453.0 MB
  float*          resb  = (float*)(ws + off);            off += (size_t)Q_  * H_  * 4;  //  75.5 MB
  unsigned short* hbuf  = (unsigned short*)(ws + off);   off += (size_t)Q_  * H_  * 2;  //  37.7 MB

  float* out  = (float*)d_out;
  float* hfin = out + (size_t)B_ * PR_ * PC_ * H_;

  const int conv_elems = NG_ * E_ + NG_ * GH_ + H_ * GH_ + H_ * E_ + Q_ * E_;
  prep_convert<<<(conv_elems + 255) / 256, 256, 0, stream>>>(
      x, W_ih, W_hh, W_comp, W_res, wih, whh, wcomp, wres, xbf, progress);

  // 4608 m-tiles x (12 xg-segments + 2 res-segments)
  xg_res_gemm<<<(Q_ / 16) * 14, 256, 0, stream>>>(
      xbf, b_ih, b_hh, b_comp, b_res, wih, wres, xg, resb);

  gru2d_wavefront<<<PR_, 256, 0, stream>>>(
      b_hh, gamma, beta, whh, wcomp, xg, resb, hbuf, out, progress);

  finalize_hfinal<<<(B_ * H_ + 255) / 256, 256, 0, stream>>>(out, hfin);
}
